// CrossCovarianceAttn_60730837565763
// MI455X (gfx1250) — matrix operand order repaired and checked by reading, not compiled, not run on hardware
//
#include <hip/hip_runtime.h>
#include <hip/hip_bf16.h>

#define BATCH   4
#define SEQ     8192
#define DIM     768
#define NHEADS  12
#define HD      64
#define C3      (3 * DIM)
#define MROWS   (BATCH * SEQ)   // 32768
#define QKCOLS  (2 * DIM)       // 1536 q+k channels per batch

typedef __attribute__((ext_vector_type(16))) __bf16 bf16x16;
typedef __attribute__((ext_vector_type(8)))  __bf16 bf16x8;
typedef __attribute__((ext_vector_type(8)))  float  floatx8;

// ---------- bf16 helpers (RNE via integer rounding) ----------
__device__ __forceinline__ __bf16 f2bf(float f) {
    unsigned u = __builtin_bit_cast(unsigned, f);
    unsigned r = u + 0x7FFFu + ((u >> 16) & 1u);
    unsigned short h = (unsigned short)(r >> 16);
    return __builtin_bit_cast(__bf16, h);
}
__device__ __forceinline__ float bf2f(__bf16 b) {
    unsigned short h = __builtin_bit_cast(unsigned short, b);
    unsigned u = ((unsigned)h) << 16;
    return __builtin_bit_cast(float, u);
}

// ---------- WMMA wrapper ----------
__device__ __forceinline__ floatx8 wmma_bf16(bf16x16 a, bf16x16 b, floatx8 c) {
    return __builtin_amdgcn_wmma_f32_16x16x32_bf16(
        false, a, false, b, (short)0, c, false, false);
}

// ---------- vectorized fragment loaders (ISA 7.12.2 layouts, wave32) ----------
// A 16x32 (MxK), row-major A[m,k] = p[m*lda + k].
// Lane needs a[0..7] = A[m, half*8 + 0..7], a[8..15] = A[m, 16 + half*8 + 0..7]
// -> two contiguous 16B chunks -> two b128 loads.
__device__ __forceinline__ bf16x16 load_a_row(const __bf16* p, int lda, int lane) {
    const int half = lane >> 4;
    const int m    = lane & 15;
    const __bf16* q = p + (size_t)m * lda + (half << 3);
    bf16x8 lo = *(const bf16x8*)(q);
    bf16x8 hi = *(const bf16x8*)(q + 16);
    return __builtin_shufflevector(lo, hi, 0, 1, 2, 3, 4, 5, 6, 7,
                                   8, 9, 10, 11, 12, 13, 14, 15);
}
// B 32x16 (KxN) stored K-major: B[k,n] = p[n*ldk + k].
// Lane needs b[0..7] = B[half*8 + 0..7, n], b[8..15] = B[16 + half*8 + 0..7, n] -> two b128 loads.
__device__ __forceinline__ bf16x16 load_b_kmajor(const __bf16* p, int ldk, int lane) {
    const int half = lane >> 4;
    const int n    = lane & 15;
    const __bf16* q = p + (size_t)n * ldk + (half << 3);
    bf16x8 lo = *(const bf16x8*)(q);
    bf16x8 hi = *(const bf16x8*)(q + 16);
    return __builtin_shufflevector(lo, hi, 0, 1, 2, 3, 4, 5, 6, 7,
                                   8, 9, 10, 11, 12, 13, 14, 15);
}

// ---------- kernel 1a: fp32 -> bf16 (plain) ----------
__global__ __launch_bounds__(256) void cvt_f32_bf16(const float* __restrict__ in,
                                                    __bf16* __restrict__ out, int n) {
    int i = blockIdx.x * blockDim.x + threadIdx.x;
    int stride = gridDim.x * blockDim.x;
    for (; i < n; i += stride) out[i] = f2bf(in[i]);
}

// ---------- kernel 1b: fp32 [K x N] -> bf16 transposed [N x K] (tiled LDS) ------
__global__ __launch_bounds__(256) void cvt_transpose(const float* __restrict__ in,
                                                     __bf16* __restrict__ out,
                                                     int K, int N) {
    __shared__ float tile[32][33];
    const int n0 = blockIdx.x * 32;
    const int k0 = blockIdx.y * 32;
    const int tx = threadIdx.x & 31;
    const int ty = threadIdx.x >> 5;  // 0..7
#pragma unroll
    for (int i = 0; i < 32; i += 8)
        tile[ty + i][tx] = in[(size_t)(k0 + ty + i) * N + n0 + tx];
    __syncthreads();
#pragma unroll
    for (int i = 0; i < 32; i += 8)
        out[(size_t)(n0 + ty + i) * K + k0 + tx] = f2bf(tile[tx][ty + i]);
}

// ---------- kernel 2: QKV GEMM, C bf16 = A @ W^T + bias ----------
// block = 128 threads = 4 waves; block tile 128(M) x 64(N); wave tile 32 x 64
__global__ __launch_bounds__(128) void gemm_bf16_out(
    const __bf16* __restrict__ A, const __bf16* __restrict__ Wt,
    const float* __restrict__ bias, __bf16* __restrict__ Cm,
    int N, int K) {
    const int lane = threadIdx.x & 31;
    const int wave = threadIdx.x >> 5;
    const int n0 = blockIdx.x * 64;
    const int m0 = blockIdx.y * 128 + wave * 32;
    floatx8 acc[2][4] = {};
    for (int kk = 0; kk < K; kk += 32) {
        bf16x16 af0 = load_a_row(A + (size_t)m0 * K + kk, K, lane);
        bf16x16 af1 = load_a_row(A + (size_t)(m0 + 16) * K + kk, K, lane);
#pragma unroll
        for (int t = 0; t < 4; ++t) {
            bf16x16 bf = load_b_kmajor(Wt + (size_t)(n0 + 16 * t) * K + kk, K, lane);
            acc[0][t] = wmma_bf16(af0, bf, acc[0][t]);
            acc[1][t] = wmma_bf16(af1, bf, acc[1][t]);
        }
    }
    const int half = lane >> 4;
    const int nn = lane & 15;
#pragma unroll
    for (int t = 0; t < 4; ++t) {
        int col = n0 + 16 * t + nn;
        float bv = bias[col];
#pragma unroll
        for (int u = 0; u < 2; ++u)
#pragma unroll
            for (int r = 0; r < 8; ++r) {
                int row = m0 + 16 * u + (half << 3) + r;
                Cm[(size_t)row * N + col] = f2bf(acc[u][t][r] + bv);
            }
    }
}

// ---------- kernel 3: bf16 transpose of q,k channels: [n, c] -> [c, n] per batch --
__global__ __launch_bounds__(256) void qk_transpose(const __bf16* __restrict__ in,
                                                    __bf16* __restrict__ out) {
    __shared__ __bf16 tile[32][33];
    const int b  = blockIdx.z;
    const int c0 = blockIdx.x * 32;  // 0..1535 (q then k channels)
    const int n0 = blockIdx.y * 32;
    const int tx = threadIdx.x & 31;
    const int ty = threadIdx.x >> 5;
#pragma unroll
    for (int i = 0; i < 32; i += 8)
        tile[ty + i][tx] = in[(size_t)(b * SEQ + n0 + ty + i) * C3 + c0 + tx];
    __syncthreads();
#pragma unroll
    for (int i = 0; i < 32; i += 8)
        out[(size_t)(b * QKCOLS + c0 + ty + i) * SEQ + n0 + tx] = tile[tx][ty + i];
}

// ---------- kernel 4: per-(b, q/k channel) inverse L2 norm (x temperature for q) --
__global__ __launch_bounds__(256) void colnorm_kernel(
    const __bf16* __restrict__ QKV, const float* __restrict__ temp,
    float* __restrict__ Sinv) {
    const int col = blockIdx.x * 256 + threadIdx.x;  // 0..1535
    const int b = blockIdx.y;
    const __bf16* p = QKV + (size_t)b * SEQ * C3 + col;
    float s = 0.f;
    for (int n = 0; n < SEQ; ++n) {
        float v = bf2f(p[(size_t)n * C3]);
        s += v * v;
    }
    float inv = 1.0f / fmaxf(sqrtf(s), 1e-12f);
    if (col < DIM) inv *= temp[col / HD];  // fold temperature into q scale
    Sinv[b * QKCOLS + col] = inv;
}

// ---------- kernel 5: Gram + scale + softmax -> S (row-major [d][e]) per (b,h) ---
// block = 128 threads = 4 waves; wave w owns rows d = 16w..16w+15, all 64 e cols
__global__ __launch_bounds__(128) void attn_kernel(
    const __bf16* __restrict__ QKT, const float* __restrict__ Sinv,
    __bf16* __restrict__ S) {
    const int lane = threadIdx.x & 31;
    const int wave = threadIdx.x >> 5;
    const int bh = blockIdx.x;
    const int b = bh / NHEADS, h = bh % NHEADS;
    // q rows: QKT[(b*1536 + h*64 + d) * SEQ + n]   (row-major A, lda = SEQ)
    // k rows: QKT[(b*1536 + 768 + h*64 + e) * SEQ + n]  (K-major B, ldk = SEQ)
    const __bf16* qp = QKT + ((size_t)b * QKCOLS + h * HD + wave * 16) * SEQ;
    const __bf16* kp = QKT + ((size_t)b * QKCOLS + DIM + h * HD) * SEQ;
    floatx8 acc[4] = {};
    for (int kk = 0; kk < SEQ; kk += 32) {
        bf16x16 af = load_a_row(qp + kk, SEQ, lane);
#pragma unroll
        for (int t = 0; t < 4; ++t) {
            bf16x16 bf = load_b_kmajor(kp + (size_t)(16 * t) * SEQ + kk, SEQ, lane);
            acc[t] = wmma_bf16(af, bf, acc[t]);
        }
    }
    const int half = lane >> 4;
    const int nn = lane & 15;
    const float* sq = Sinv + b * QKCOLS + h * HD;        // indexed by d
    const float* sk = Sinv + b * QKCOLS + DIM + h * HD;  // indexed by e
    float qs[8];
#pragma unroll
    for (int r = 0; r < 8; ++r) qs[r] = sq[wave * 16 + (half << 3) + r];
#pragma unroll
    for (int t = 0; t < 4; ++t) {
        float ks = sk[16 * t + nn];
#pragma unroll
        for (int r = 0; r < 8; ++r) acc[t][r] *= qs[r] * ks;
    }
    // row softmax over 64 e-columns (row = one 16-lane half x 4 accums)
    float mx[8], sm[8];
#pragma unroll
    for (int r = 0; r < 8; ++r) {
        float m = fmaxf(fmaxf(acc[0][r], acc[1][r]), fmaxf(acc[2][r], acc[3][r]));
#pragma unroll
        for (int off = 1; off < 16; off <<= 1) m = fmaxf(m, __shfl_xor(m, off, 32));
        mx[r] = m;
    }
#pragma unroll
    for (int r = 0; r < 8; ++r) {
        float s = 0.f;
#pragma unroll
        for (int t = 0; t < 4; ++t) {
            acc[t][r] = __expf(acc[t][r] - mx[r]);
            s += acc[t][r];
        }
#pragma unroll
        for (int off = 1; off < 16; off <<= 1) s += __shfl_xor(s, off, 32);
        sm[r] = 1.0f / s;
    }
    // store S row-major [d][e]: K(e)-major for the AV GEMM's B operand
    __bf16* stp = S + (size_t)bh * HD * HD;
#pragma unroll
    for (int t = 0; t < 4; ++t) {
        int e = 16 * t + nn;
#pragma unroll
        for (int r = 0; r < 8; ++r) {
            int d = wave * 16 + (half << 3) + r;
            stp[d * HD + e] = f2bf(acc[t][r] * sm[r]);
        }
    }
}

// ---------- kernel 6: O[n,d] = V[n,e] @ S[d,e]^T per (b,h) -> [B,N,C] bf16 -------
// block = 128 threads = 4 waves; wave tile 32(n) x 64(d)
__global__ __launch_bounds__(128) void av_kernel(
    const __bf16* __restrict__ QKV, const __bf16* __restrict__ S,
    __bf16* __restrict__ AO) {
    const int lane = threadIdx.x & 31;
    const int wave = threadIdx.x >> 5;
    const int bh = blockIdx.y;
    const int b = bh / NHEADS, h = bh % NHEADS;
    const int m0 = blockIdx.x * 128 + wave * 32;  // n within sequence
    const __bf16* vp = QKV + (size_t)b * SEQ * C3 + 2 * DIM + h * HD;  // A, lda=C3
    const __bf16* sp = S + (size_t)bh * HD * HD;  // B[k=e,n=d] = sp[d*64 + e]
    floatx8 acc[2][4] = {};
#pragma unroll
    for (int kk = 0; kk < HD; kk += 32) {
        bf16x16 af0 = load_a_row(vp + (size_t)m0 * C3 + kk, C3, lane);
        bf16x16 af1 = load_a_row(vp + (size_t)(m0 + 16) * C3 + kk, C3, lane);
#pragma unroll
        for (int t = 0; t < 4; ++t) {
            bf16x16 bf = load_b_kmajor(sp + (16 * t) * HD + kk, HD, lane);
            acc[0][t] = wmma_bf16(af0, bf, acc[0][t]);
            acc[1][t] = wmma_bf16(af1, bf, acc[1][t]);
        }
    }
    const int half = lane >> 4;
    const int nn = lane & 15;
    __bf16* op = AO + (size_t)b * SEQ * DIM + h * HD;
#pragma unroll
    for (int t = 0; t < 4; ++t) {
        int d = 16 * t + nn;
#pragma unroll
        for (int u = 0; u < 2; ++u)
#pragma unroll
            for (int r = 0; r < 8; ++r) {
                int row = m0 + 16 * u + (half << 3) + r;
                op[(size_t)row * DIM + d] = f2bf(acc[u][t][r]);
            }
    }
}

// ---------- kernel 7: projection GEMM, fp32 out + bias ----------
__global__ __launch_bounds__(128) void gemm_f32_out(
    const __bf16* __restrict__ A, const __bf16* __restrict__ Wt,
    const float* __restrict__ bias, float* __restrict__ Cm,
    int N, int K) {
    const int lane = threadIdx.x & 31;
    const int wave = threadIdx.x >> 5;
    const int n0 = blockIdx.x * 64;
    const int m0 = blockIdx.y * 128 + wave * 32;
    floatx8 acc[2][4] = {};
    for (int kk = 0; kk < K; kk += 32) {
        bf16x16 af0 = load_a_row(A + (size_t)m0 * K + kk, K, lane);
        bf16x16 af1 = load_a_row(A + (size_t)(m0 + 16) * K + kk, K, lane);
#pragma unroll
        for (int t = 0; t < 4; ++t) {
            bf16x16 bf = load_b_kmajor(Wt + (size_t)(n0 + 16 * t) * K + kk, K, lane);
            acc[0][t] = wmma_bf16(af0, bf, acc[0][t]);
            acc[1][t] = wmma_bf16(af1, bf, acc[1][t]);
        }
    }
    const int half = lane >> 4;
    const int nn = lane & 15;
#pragma unroll
    for (int t = 0; t < 4; ++t) {
        int col = n0 + 16 * t + nn;
        float bv = bias[col];
#pragma unroll
        for (int u = 0; u < 2; ++u)
#pragma unroll
            for (int r = 0; r < 8; ++r) {
                int row = m0 + 16 * u + (half << 3) + r;
                Cm[(size_t)row * N + col] = acc[u][t][r] + bv;
            }
    }
}

extern "C" void kernel_launch(void* const* d_in, const int* in_sizes, int n_in,
                              void* d_out, int out_size, void* d_ws, size_t ws_size,
                              hipStream_t stream) {
    (void)in_sizes; (void)n_in; (void)out_size; (void)ws_size;
    const float* x     = (const float*)d_in[0];
    const float* Wqkv  = (const float*)d_in[1];
    const float* bqkv  = (const float*)d_in[2];
    const float* temp  = (const float*)d_in[3];
    const float* Wproj = (const float*)d_in[4];
    const float* bproj = (const float*)d_in[5];
    float* out = (float*)d_out;

    char* ws = (char*)d_ws;
    size_t off = 0;
    auto alloc = [&](size_t bytes) -> void* {
        void* p = ws + off;
        off = (off + bytes + 255) & ~(size_t)255;
        return p;
    };
    __bf16* Xb     = (__bf16*)alloc((size_t)MROWS * DIM * 2);
    __bf16* WqkvT  = (__bf16*)alloc((size_t)DIM * C3 * 2);    // [2304 x 768] K-major
    __bf16* WprojT = (__bf16*)alloc((size_t)DIM * DIM * 2);   // [768 x 768] K-major
    __bf16* QKVb   = (__bf16*)alloc((size_t)MROWS * C3 * 2);
    __bf16* QKT    = (__bf16*)alloc((size_t)BATCH * QKCOLS * SEQ * 2);
    __bf16* AOb    = (__bf16*)alloc((size_t)MROWS * DIM * 2);
    float*  Sinv   = (float*)alloc((size_t)BATCH * QKCOLS * 4);
    __bf16* Sbuf   = (__bf16*)alloc((size_t)BATCH * NHEADS * HD * HD * 2);

    // 1) converts (weights transposed to K-major for contiguous B fragments)
    cvt_f32_bf16<<<2048, 256, 0, stream>>>(x, Xb, MROWS * DIM);
    cvt_transpose<<<dim3(C3 / 32, DIM / 32), 256, 0, stream>>>(Wqkv, WqkvT, DIM, C3);
    cvt_transpose<<<dim3(DIM / 32, DIM / 32), 256, 0, stream>>>(Wproj, WprojT, DIM, DIM);

    // 2) qkv = x @ Wqkv + bqkv   [32768 x 2304], K = 768
    gemm_bf16_out<<<dim3(C3 / 64, MROWS / 128), 128, 0, stream>>>(
        Xb, WqkvT, bqkv, QKVb, C3, DIM);

    // 3) transpose q,k channels to [b, channel, N] for contiguous Gram fragments
    qk_transpose<<<dim3(QKCOLS / 32, SEQ / 32, BATCH), 256, 0, stream>>>(QKVb, QKT);

    // 4) per-channel inverse norms (temperature folded into q)
    colnorm_kernel<<<dim3(QKCOLS / 256, BATCH), 256, 0, stream>>>(QKVb, temp, Sinv);

    // 5) scaled Gram + softmax -> S per (b,h)
    attn_kernel<<<BATCH * NHEADS, 128, 0, stream>>>(QKT, Sinv, Sbuf);

    // 6) O = V @ S^T, written directly into [B,N,C] layout
    av_kernel<<<dim3(SEQ / 128, BATCH * NHEADS), 128, 0, stream>>>(QKVb, Sbuf, AOb);

    // 7) out = AO @ Wproj + bproj   [32768 x 768], K = 768, fp32 out
    gemm_f32_out<<<dim3(DIM / 64, MROWS / 128), 128, 0, stream>>>(
        AOb, WprojT, bproj, out, DIM, DIM);
}
